// GraphAttentionLayer_16277926052537
// MI455X (gfx1250) — compile-verified, matrix operand order checked by reading, not run on hardware
//
#include <hip/hip_runtime.h>
#include <stdint.h>

// ---- problem constants (match reference) ----
constexpr int NN     = 4096;
constexpr int IN_F   = 512;
constexpr int OUT_F  = 256;
constexpr int HEADS  = 4;
constexpr float ALPHA   = 0.2f;
constexpr float NEG_INF = -1e18f;

typedef __attribute__((ext_vector_type(16))) __bf16 v16bf;
typedef __attribute__((ext_vector_type(8)))  float  v8f;

union BF16Frag {
  v16bf        v;
  __bf16       b[16];
  unsigned int u[8];
};

// f32 pair -> packed bf16 dword via HW cvt (lo = first K of the pair)
__device__ __forceinline__ unsigned int packbf2(float lo, float hi) {
  union { __bf16 b[2]; unsigned int u; } t;
  t.b[0] = (__bf16)lo;
  t.b[1] = (__bf16)hi;
  return t.u;
}

__device__ __forceinline__ v8f wmma_bf16(const BF16Frag& A, const BF16Frag& B, v8f c) {
  return __builtin_amdgcn_wmma_f32_16x16x32_bf16(false, A.v, false, B.v,
                                                 (short)0, c, false, false);
}

// load 16 f32 (two 8-float groups: k = base..base+7 and base+16..base+23)
// and convert into a WMMA 16-bit operand fragment using HW bf16 converts
__device__ __forceinline__ void load_frag_f32(BF16Frag& F, const float* p) {
  float4 x0 = *(const float4*)(p);
  float4 x1 = *(const float4*)(p + 4);
  float4 x2 = *(const float4*)(p + 16);
  float4 x3 = *(const float4*)(p + 20);
  F.b[0]  = (__bf16)x0.x; F.b[1]  = (__bf16)x0.y;
  F.b[2]  = (__bf16)x0.z; F.b[3]  = (__bf16)x0.w;
  F.b[4]  = (__bf16)x1.x; F.b[5]  = (__bf16)x1.y;
  F.b[6]  = (__bf16)x1.z; F.b[7]  = (__bf16)x1.w;
  F.b[8]  = (__bf16)x2.x; F.b[9]  = (__bf16)x2.y;
  F.b[10] = (__bf16)x2.z; F.b[11] = (__bf16)x2.w;
  F.b[12] = (__bf16)x3.x; F.b[13] = (__bf16)x3.y;
  F.b[14] = (__bf16)x3.z; F.b[15] = (__bf16)x3.w;
}

// ---------------------------------------------------------------------------
// K1: Wh[h][n][f] = sum_k h[n][k] * W[h][f][k]   (bf16 WMMA, f32 accumulate)
// Also emits WhT[h][f][n] as packed bf16 pairs for K4's B operands.
// Block = 256 thr (8 waves); block handles (head, 16-row strip); wave owns 32 cols.
// ---------------------------------------------------------------------------
__global__ __launch_bounds__(256) void k1_gemm(const float* __restrict__ h,
                                               const float* __restrict__ W,
                                               float* __restrict__ Wh,
                                               unsigned int* __restrict__ WhT) {
  const int blk   = blockIdx.x;          // head*256 + strip
  const int head  = blk >> 8;
  const int strip = blk & 255;
  const int lane  = threadIdx.x & 31;
  const int wave  = threadIdx.x >> 5;
  const int hi    = lane >> 4;
  const int kb    = hi * 8;
  const int row   = strip * 16 + (lane & 15);

  const float* hrow = h + row * IN_F;
  const int f0 = wave * 32 + (lane & 15);
  const int f1 = f0 + 16;
  const float* w0 = W + (head * OUT_F + f0) * IN_F;
  const float* w1 = W + (head * OUT_F + f1) * IN_F;

  v8f c0 = {}; v8f c1 = {};
  for (int k0 = 0; k0 < IN_F; k0 += 32) {
    BF16Frag A, B0, B1;
    load_frag_f32(A,  hrow + k0 + kb);
    load_frag_f32(B0, w0   + k0 + kb);
    load_frag_f32(B1, w1   + k0 + kb);
    c0 = wmma_bf16(A, B0, c0);
    c1 = wmma_bf16(A, B1, c1);
  }

  // D tile: lane holds column f0/f1, VGPR v -> row (v + 8*hi) of the strip
  float* whb = Wh + ((head * NN) + strip * 16) * OUT_F;
  for (int v = 0; v < 8; ++v) {
    int r = v + 8 * hi;
    whb[r * OUT_F + f0] = c0[v];
    whb[r * OUT_F + f1] = c1[v];
  }
  // WhT bf16: rows v=0..7 are consecutive -> pack pairs
  unsigned int* t0 = WhT + (head * OUT_F + f0) * (NN / 2) + strip * 8 + 4 * hi;
  unsigned int* t1 = WhT + (head * OUT_F + f1) * (NN / 2) + strip * 8 + 4 * hi;
  for (int p = 0; p < 4; ++p) {
    t0[p] = packbf2(c0[2 * p], c0[2 * p + 1]);
    t1[p] = packbf2(c1[2 * p], c1[2 * p + 1]);
  }
}

// ---------------------------------------------------------------------------
// K2: s1[h][i] = Wh[h][i][:] . a[h][:F] ; s2 likewise with a[h][F:]
// one wave per (h, i)
// ---------------------------------------------------------------------------
__global__ __launch_bounds__(256) void k2_scores(const float* __restrict__ Wh,
                                                 const float* __restrict__ a,
                                                 float* __restrict__ s1,
                                                 float* __restrict__ s2) {
  const int g    = blockIdx.x * 8 + (threadIdx.x >> 5);   // 0 .. H*N-1
  const int lane = threadIdx.x & 31;
  const int head = g >> 12;
  const int i    = g & (NN - 1);
  const float* wr = Wh + (head * NN + i) * OUT_F;
  const float* av = a + head * (2 * OUT_F);
  float x1 = 0.f, x2 = 0.f;
  for (int f = lane; f < OUT_F; f += 32) {
    float w = wr[f];
    x1 += w * av[f];
    x2 += w * av[OUT_F + f];
  }
  for (int o = 16; o > 0; o >>= 1) {
    x1 += __shfl_xor(x1, o);
    x2 += __shfl_xor(x2, o);
  }
  if (lane == 0) { s1[g] = x1; s2[g] = x2; }
}

// ---------------------------------------------------------------------------
// K3: per (h, i): masked row max, then sum of exp(e - max). One wave per row.
// ---------------------------------------------------------------------------
__global__ __launch_bounds__(256) void k3_softstats(const int* __restrict__ adj,
                                                    const float* __restrict__ s1,
                                                    const float* __restrict__ s2,
                                                    float* __restrict__ rmax,
                                                    float* __restrict__ rsum) {
  const int g    = blockIdx.x * 8 + (threadIdx.x >> 5);
  const int lane = threadIdx.x & 31;
  const int head = g >> 12;
  const int i    = g & (NN - 1);
  const int*   arow = adj + i * NN;
  const float* s2h  = s2 + head * NN;
  const float  s1i  = s1[g];

  float mx = NEG_INF;
  for (int j = lane; j < NN; j += 32) {
    float e = s1i + s2h[j];
    e = e > 0.f ? e : ALPHA * e;
    e = arow[j] > 0 ? e : NEG_INF;
    mx = fmaxf(mx, e);
  }
  for (int o = 16; o > 0; o >>= 1) mx = fmaxf(mx, __shfl_xor(mx, o));

  float sum = 0.f;
  for (int j = lane; j < NN; j += 32) {
    float e = s1i + s2h[j];
    e = e > 0.f ? e : ALPHA * e;
    e = arow[j] > 0 ? e : NEG_INF;
    sum += __expf(e - mx);
  }
  for (int o = 16; o > 0; o >>= 1) sum += __shfl_xor(sum, o);
  if (lane == 0) { rmax[g] = mx; rsum[g] = sum; }
}

// ---------------------------------------------------------------------------
// K4: out = elu( softmax(masked e) @ Wh ), heads interleaved [N, H*F].
// Block = (head, 16-row strip). P tile (16x32) built cooperatively in LDS in
// the exact WMMA A-fragment layout; double-buffered so only ONE barrier per
// K-step is needed (next write to a buffer is 2 iterations + 1 barrier away).
// Each wave owns a 32-col output slice.
// ---------------------------------------------------------------------------
__global__ __launch_bounds__(256) void k4_attn(const int* __restrict__ adj,
                                               const float* __restrict__ s1,
                                               const float* __restrict__ s2,
                                               const float* __restrict__ rmax,
                                               const float* __restrict__ rsum,
                                               const unsigned int* __restrict__ WhT,
                                               float* __restrict__ out) {
  __shared__ unsigned int Ptile[2][256];  // 2 x (32 lane-slots x 8 dwords)
  const int blk   = blockIdx.x;
  const int head  = blk >> 8;
  const int strip = blk & 255;
  const int t     = threadIdx.x;
  const int lane  = t & 31;
  const int wave  = t >> 5;

  // ---- producer role: thread t fills one A-fragment dword per K-step ----
  const int pl  = t >> 3;                 // fragment lane slot 0..31
  const int pv  = t & 7;                  // fragment dword 0..7
  const int pkb = (pl >> 4) * 8;
  const int pk  = pkb + (pv < 4 ? 2 * pv : 16 + 2 * (pv - 4));
  const int pi  = strip * 16 + (pl & 15); // global row
  const float ps1 = s1[head * NN + pi];
  const float pmx = rmax[head * NN + pi];
  const int*   parow = adj + pi * NN;
  const float* s2h   = s2 + head * NN;

  // ---- consumer role ----
  const int hi = lane >> 4;
  const int kb = hi * 8;
  const int f0 = wave * 32 + (lane & 15);
  const int f1 = f0 + 16;
  const unsigned int* bt0 = WhT + (head * OUT_F + f0) * (NN / 2);
  const unsigned int* bt1 = WhT + (head * OUT_F + f1) * (NN / 2);

  v8f c0 = {}; v8f c1 = {};
#pragma unroll 2
  for (int jb = 0; jb < NN; jb += 32) {
    const int buf = (jb >> 5) & 1;
    // produce P = exp(e - rowmax); masked entries underflow to 0
    {
      int j = jb + pk;
      float e0 = ps1 + s2h[j];
      e0 = e0 > 0.f ? e0 : ALPHA * e0;
      e0 = parow[j] > 0 ? e0 : NEG_INF;
      float e1 = ps1 + s2h[j + 1];
      e1 = e1 > 0.f ? e1 : ALPHA * e1;
      e1 = parow[j + 1] > 0 ? e1 : NEG_INF;
      Ptile[buf][pl * 8 + pv] = packbf2(__expf(e0 - pmx), __expf(e1 - pmx));
    }
    __syncthreads();   // publish buf; also fences prior reads of this buffer

    BF16Frag A, B0, B1;
    {
      const uint4* lp = (const uint4*)&Ptile[buf][lane * 8];  // 16B aligned
      uint4 a0 = lp[0];
      uint4 a1 = lp[1];
      A.u[0]=a0.x; A.u[1]=a0.y; A.u[2]=a0.z; A.u[3]=a0.w;
      A.u[4]=a1.x; A.u[5]=a1.y; A.u[6]=a1.z; A.u[7]=a1.w;
    }
    {
      const int base = (jb >> 1) + (kb >> 1);
      uint4 x0 = *(const uint4*)(bt0 + base);
      uint4 x1 = *(const uint4*)(bt0 + base + 8);
      B0.u[0]=x0.x; B0.u[1]=x0.y; B0.u[2]=x0.z; B0.u[3]=x0.w;
      B0.u[4]=x1.x; B0.u[5]=x1.y; B0.u[6]=x1.z; B0.u[7]=x1.w;
      uint4 y0 = *(const uint4*)(bt1 + base);
      uint4 y1 = *(const uint4*)(bt1 + base + 8);
      B1.u[0]=y0.x; B1.u[1]=y0.y; B1.u[2]=y0.z; B1.u[3]=y0.w;
      B1.u[4]=y1.x; B1.u[5]=y1.y; B1.u[6]=y1.z; B1.u[7]=y1.w;
    }
    c0 = wmma_bf16(A, B0, c0);
    c1 = wmma_bf16(A, B1, c1);
  }

  // epilogue: normalize by row sum, ELU, store [N, H*F]
  for (int v = 0; v < 8; ++v) {
    int row = strip * 16 + v + 8 * hi;
    float Z = rsum[head * NN + row];
    float o0 = c0[v] / Z;
    float o1 = c1[v] / Z;
    o0 = o0 > 0.f ? o0 : (__expf(o0) - 1.f);
    o1 = o1 > 0.f ? o1 : (__expf(o1) - 1.f);
    float* orow = out + row * (HEADS * OUT_F) + head * OUT_F;
    orow[f0] = o0;
    orow[f1] = o1;
  }
}

// ---------------------------------------------------------------------------
extern "C" void kernel_launch(void* const* d_in, const int* in_sizes, int n_in,
                              void* d_out, int out_size, void* d_ws, size_t ws_size,
                              hipStream_t stream) {
  const float* h   = (const float*)d_in[0];   // [N, IN_F]
  const int*   adj = (const int*)  d_in[1];   // [N, N]
  const float* W   = (const float*)d_in[2];   // [H, OUT_F, IN_F]
  const float* a   = (const float*)d_in[3];   // [H, 2*OUT_F]
  float* out = (float*)d_out;                 // [N, H*OUT_F]

  char* ws = (char*)d_ws;
  float*        Wh   = (float*)ws;                                  // 16 MiB
  unsigned int* WhT  = (unsigned int*)(ws + (size_t)16777216);      //  8 MiB
  float*        s1   = (float*)(ws + (size_t)25165824);
  float*        s2   = (float*)(ws + (size_t)25231360);
  float*        rmax = (float*)(ws + (size_t)25296896);
  float*        rsum = (float*)(ws + (size_t)25362432);

  k1_gemm     <<<HEADS * (NN / 16), 256, 0, stream>>>(h, W, Wh, WhT);
  k2_scores   <<<(HEADS * NN) / 8,  256, 0, stream>>>(Wh, a, s1, s2);
  k3_softstats<<<(HEADS * NN) / 8,  256, 0, stream>>>(adj, s1, s2, rmax, rsum);
  k4_attn     <<<HEADS * (NN / 16), 256, 0, stream>>>(adj, s1, s2, rmax, rsum, WhT, out);
}
